// QuantizeEMAReset_27693949125325
// MI455X (gfx1250) — matrix operand order repaired and checked by reading, not compiled, not run on hardware
//
#include <hip/hip_runtime.h>
#include <hip/hip_bf16.h>

// ---------------------------------------------------------------------------
// QuantizeEMAReset for MI455X (gfx1250, wave32, WMMA)
// Outputs: x_d_out (N*C*T floats), commit_loss, perplexity
// code_sum / code_count inputs are dead w.r.t. the outputs (new_codebook is
// deleted in the reference), so they are ignored.
// ---------------------------------------------------------------------------

typedef __attribute__((ext_vector_type(16))) _Float16 v16h;
typedef __attribute__((ext_vector_type(8)))  _Float16 v8h;
typedef __attribute__((ext_vector_type(8)))  float    v8f;

constexpr int NN = 256;   // batch
constexpr int CC = 512;   // channels (= K of GEMM = code_dim)
constexpr int TT = 256;   // time
constexpr int NB = 512;   // nb_code (= N of GEMM)
constexpr int M_ROWS = NN * TT;          // 65536 flattened rows

// k_argmin tiling
constexpr int ROWS_PER_BLK = 128;        // 8 waves * 16 rows
constexpr int SA = 528;                  // ldsA stride (f16), 1056B rows: 16B-aligned, conflict-free
constexpr int SB = 136;                  // sizes the B region only; ldsB rows use stride SA
constexpr int A_BYTES = ROWS_PER_BLK * SA * 2;   // 135168
constexpr int B_BYTES = NB * SB * 2;             // 139264 (code-major chunk of 128 codes)
constexpr int SMEM1 = A_BYTES + B_BYTES;         // 274432 <= 320KB/WGP

// k_gather tiling
constexpr int TGT = 64;                  // t-tile
constexpr int SG = 513;                  // row stride (floats), conflict-free transpose
constexpr int SMEM2 = TGT * SG * 4;      // 131328

// ---------------------------------------------------------------------------
// Prep: zero histogram, compute ||c_j||^2 in fp32 (exact part of distance)
// ---------------------------------------------------------------------------
__global__ void __launch_bounds__(256) k_prep(const float* __restrict__ cb,
                                              unsigned* __restrict__ counts,
                                              float* __restrict__ cnorm)
{
    int g = blockIdx.x * 256 + threadIdx.x;
    if (g < NB) {
        counts[g] = 0u;
        const float* row = cb + (size_t)g * CC;
        float s = 0.f;
        for (int c = 0; c < CC; ++c) { float v = row[c]; s += v * v; }
        cnorm[g] = s;
    }
}

// ---------------------------------------------------------------------------
// Argmin over codes via v_wmma_f32_16x16x32_f16.
//   score(i,j) = ||c_j||^2 - 2 * <x_i, c_j>   (||x_i||^2 is row-constant)
// Block: 256 threads (8 waves), 128 rows. Wave handles 16 rows x all 512 codes.
// ---------------------------------------------------------------------------
__global__ void __launch_bounds__(256, 1) k_argmin(
    const float* __restrict__ x, const float* __restrict__ cb,
    const float* __restrict__ cnorm, unsigned* __restrict__ counts,
    int* __restrict__ idx_out)
{
    extern __shared__ char smem[];
    _Float16* ldsA = (_Float16*)smem;               // [128][SA]  row-major (t, c)
    _Float16* ldsB = (_Float16*)(smem + A_BYTES);   // [128][SA]  code-major (code, c)

    const int tid  = threadIdx.x;
    const int lane = tid & 31;
    const int wave = tid >> 5;
    const int half = lane >> 4;     // 0: lanes 0-15, 1: lanes 16-31
    const int l16  = lane & 15;

    const int i0 = blockIdx.x * ROWS_PER_BLK;       // flat row base (n*T + t)
    const int n  = i0 >> 8;                         // T = 256, 128 | 256
    const int t0 = i0 & 255;
    const float* xn = x + (size_t)n * (CC * TT);

    // Stage A: x[n][c][t0+tt] -> f16 ldsA[tt][c]  (coalesced global, conflict-free LDS)
    for (int e = tid; e < CC * ROWS_PER_BLK; e += 256) {
        int c = e >> 7, tt = e & 127;
        ldsA[tt * SA + c] = (_Float16)xn[c * TT + t0 + tt];
    }
    __syncthreads();

    // A fragments in registers: 16 K-tiles (K=32 each).
    // ISA 16-bit A layout: lane<16 -> row=lane, K {kt*32+0..7, +16..23};
    //                      lane>=16 -> row=lane-16, K {+8..15, +24..31}.
    v16h afrag[16];
    {
        const _Float16* abase = ldsA + (wave * 16 + l16) * SA + half * 8;
        #pragma unroll
        for (int kt = 0; kt < 16; ++kt) {
            v8h lo = *(const v8h*)(abase + kt * 32);        // 16B ds_load_b128
            v8h hi = *(const v8h*)(abase + kt * 32 + 16);
            afrag[kt] = __builtin_shufflevector(lo, hi,
                0, 1, 2, 3, 4, 5, 6, 7, 8, 9, 10, 11, 12, 13, 14, 15);
        }
    }

    float bestV[8];
    int   bestI[8];
    #pragma unroll
    for (int r = 0; r < 8; ++r) { bestV[r] = 3.4e38f; bestI[r] = 0; }

    // Codebook streamed in 4 chunks of 128 codes, staged code-major in LDS,
    // shared by all 8 waves of the block.
    for (int chunk = 0; chunk < 4; ++chunk) {
        __syncthreads();   // prior chunk's B consumers done
        const float* cbc = cb + (size_t)chunk * 128 * CC;
        for (int e = tid; e < 128 * CC; e += 256) {
            int code = e >> 9, c = e & 511;                 // consecutive tid -> consecutive c
            ldsB[code * SA + c] = (_Float16)cbc[code * CC + c];
        }
        __syncthreads();

        // Prefetch next chunk (256KB) toward cache while this chunk computes
        // (emits gfx1250 global_prefetch_b8; no counters, fire-and-forget).
        if (chunk < 3) {
            const char* nxt = (const char*)(cb + (size_t)(chunk + 1) * 128 * CC);
            #pragma unroll
            for (int p = 0; p < 4; ++p)
                __builtin_prefetch(nxt + ((size_t)tid * 4 + p) * 256, 0, 1);
        }

        #pragma unroll
        for (int nt = 0; nt < 8; ++nt) {                    // 16-code tiles in chunk
            v8f acc = {0.f, 0.f, 0.f, 0.f, 0.f, 0.f, 0.f, 0.f};
            // B layout: lane holds N = l16, K {kt*32 + 8*half + 0..7, +16..23} (same as A).
            // Two bases so every ds_load_b128 below uses an immediate offset
            // (max 7*64 + 32 + 16 < 64KB) -> no address VALU in the hot loop,
            // no WMMA WAR-hazard nops.
            const _Float16* bbL = ldsB + (nt * 16 + l16) * SA + half * 8;
            const _Float16* bbH = bbL + 8 * 32;

            // Depth-1 software pipeline: issue kt+1 fragment loads before the
            // WMMA that consumes kt, hiding LDS latency behind the matrix op.
            // (Deeper pipelining spills: B operands must stay in VGPRs.)
            v8h lo = *(const v8h*)(bbL);
            v8h hi = *(const v8h*)(bbL + 16);
            #pragma unroll
            for (int kt = 0; kt < 16; ++kt) {
                v8h nlo = lo, nhi = hi;
                if (kt < 15) {
                    const _Float16* p = (kt + 1 < 8)
                        ? (bbL + (kt + 1) * 32)
                        : (bbH + (kt + 1 - 8) * 32);
                    nlo = *(const v8h*)(p);
                    nhi = *(const v8h*)(p + 16);
                }
                v16h bfrag = __builtin_shufflevector(lo, hi,
                    0, 1, 2, 3, 4, 5, 6, 7, 8, 9, 10, 11, 12, 13, 14, 15);
                acc = __builtin_amdgcn_wmma_f32_16x16x32_f16(
                    false, afrag[kt], false, bfrag, (short)0, acc, false, false);
                lo = nlo; hi = nhi;
            }
            // C/D layout: VGPR r -> M = r + 8*half, N = l16
            const int   ncode = chunk * 128 + nt * 16 + l16;
            const float cn    = cnorm[ncode];
            #pragma unroll
            for (int r = 0; r < 8; ++r) {
                float score = cn - 2.0f * acc[r];
                if (score < bestV[r]) { bestV[r] = score; bestI[r] = ncode; }
            }
        }
    }

    // Argmin across the 16 N-lanes of each half (first-index tie-break == argmax(-d))
    #pragma unroll
    for (int off = 8; off >= 1; off >>= 1) {
        #pragma unroll
        for (int r = 0; r < 8; ++r) {
            float ov = __shfl_xor(bestV[r], off, 32);
            int   oi = __shfl_xor(bestI[r], off, 32);
            if (ov < bestV[r] || (ov == bestV[r] && oi < bestI[r])) {
                bestV[r] = ov; bestI[r] = oi;
            }
        }
    }
    if (l16 == 0) {                       // lane 0 -> rows m=0..7, lane 16 -> rows 8..15
        const int rowbase = i0 + wave * 16 + half * 8;
        #pragma unroll
        for (int r = 0; r < 8; ++r) {
            idx_out[rowbase + r] = bestI[r];
            atomicAdd(&counts[bestI[r]], 1u);
        }
    }
}

// ---------------------------------------------------------------------------
// Gather selected codebook rows through LDS (transpose) -> coalesced (N,C,T)
// output write; deterministic per-block squared-error partials.
// ---------------------------------------------------------------------------
__global__ void __launch_bounds__(256, 1) k_gather(
    const float* __restrict__ x, const float* __restrict__ cb,
    const int* __restrict__ idx, float* __restrict__ out,
    float* __restrict__ partial)
{
    extern __shared__ char smem[];
    float* rows = (float*)smem;          // [TGT][SG]
    __shared__ int   sidx[TGT];
    __shared__ float red[8];

    const int tid = threadIdx.x;
    const int blk = blockIdx.x;          // NN * (TT/TGT) = 1024 blocks
    const int n   = blk >> 2;
    const int t0  = (blk & 3) * TGT;

    if (tid < TGT) sidx[tid] = idx[n * TT + t0 + tid];
    __syncthreads();

    // Stage the 64 selected codebook rows (coalesced along c, conflict-free LDS)
    for (int e = tid; e < TGT * CC; e += 256) {
        int tt = e >> 9, c = e & 511;
        rows[tt * SG + c] = cb[(size_t)sidx[tt] * CC + c];
    }
    __syncthreads();

    const float* xn = x   + (size_t)n * (CC * TT);
    float*       on = out + (size_t)n * (CC * TT);
    float sq = 0.f;
    for (int e = tid; e < CC * TGT; e += 256) {
        int c = e >> 6, tt = e & 63;                 // consecutive tid -> consecutive t
        float cv = rows[tt * SG + c];                // stride-513: conflict-free
        float xv = xn[c * TT + t0 + tt];
        float d  = xv - cv;
        sq += d * d;
        on[c * TT + t0 + tt] = cv;                   // coalesced 256B stores
    }

    #pragma unroll
    for (int off = 16; off >= 1; off >>= 1) sq += __shfl_xor(sq, off, 32);
    if ((tid & 31) == 0) red[tid >> 5] = sq;
    __syncthreads();
    if (tid == 0) {
        float s = 0.f;
        #pragma unroll
        for (int w = 0; w < 8; ++w) s += red[w];
        partial[blk] = s;                            // deterministic (no float atomics)
    }
}

// ---------------------------------------------------------------------------
// Finalize: perplexity from histogram, commit_loss from partials. One block.
// ---------------------------------------------------------------------------
__device__ __forceinline__ float block_sum_512(float v, float* red)
{
    #pragma unroll
    for (int off = 16; off >= 1; off >>= 1) v += __shfl_xor(v, off, 32);
    __syncthreads();
    if ((threadIdx.x & 31) == 0) red[threadIdx.x >> 5] = v;
    __syncthreads();
    float t = 0.f;
    #pragma unroll
    for (int i = 0; i < 16; ++i) t += red[i];
    return t;
}

__global__ void __launch_bounds__(512) k_final(
    const unsigned* __restrict__ counts, const float* __restrict__ partial,
    float* __restrict__ out2)
{
    __shared__ float red[16];
    const int tid = threadIdx.x;           // 512 threads, one per code

    float cnt   = (float)counts[tid];
    float total = block_sum_512(cnt, red);

    float prob = cnt / (total + 1e-10f);
    float ent  = block_sum_512(prob * logf(prob + 1e-7f), red);

    float loss = block_sum_512(partial[tid] + partial[tid + 512], red);

    if (tid == 0) {
        out2[0] = loss / ((float)M_ROWS * (float)CC);   // commit_loss
        out2[1] = expf(-ent);                           // perplexity
    }
}

// ---------------------------------------------------------------------------
// Launch
// ---------------------------------------------------------------------------
extern "C" void kernel_launch(void* const* d_in, const int* in_sizes, int n_in,
                              void* d_out, int out_size, void* d_ws, size_t ws_size,
                              hipStream_t stream)
{
    (void)in_sizes; (void)n_in; (void)ws_size;
    const float* x  = (const float*)d_in[0];
    const float* cb = (const float*)d_in[1];
    // d_in[2] (code_sum) and d_in[3] (code_count) do not affect the outputs.
    float* out = (float*)d_out;

    // ws layout (4B units): counts[512] | cnorm[512] | partial[1024] | idx[65536]
    unsigned* counts  = (unsigned*)d_ws;
    float*    cnorm   = (float*)d_ws + 512;
    float*    partial = (float*)d_ws + 1024;
    int*      idxb    = (int*)d_ws + 2048;

    hipFuncSetAttribute((const void*)k_argmin,
                        hipFuncAttributeMaxDynamicSharedMemorySize, SMEM1);
    hipFuncSetAttribute((const void*)k_gather,
                        hipFuncAttributeMaxDynamicSharedMemorySize, SMEM2);

    k_prep  <<<2,                 256, 0,     stream>>>(cb, counts, cnorm);
    k_argmin<<<M_ROWS / ROWS_PER_BLK, 256, SMEM1, stream>>>(x, cb, cnorm, counts, idxb);
    k_gather<<<NN * (TT / TGT),   256, SMEM2, stream>>>(x, cb, idxb, out, partial);
    k_final <<<1,                 512, 0,     stream>>>(counts, partial, out + (out_size - 2));
}
